// SNN_78958678769973
// MI455X (gfx1250) — matrix operand order repaired and checked by reading, not compiled, not run on hardware
//
#include <hip/hip_runtime.h>
#include <hip/hip_bf16.h>

// ---------------- problem constants ----------------
#define T_STEPS 500
#define BATCH   128
#define NIN     6
#define NH      500
#define NHP     512      // padded hidden size (WMMA K multiple 32, tile multiple 16)
#define NOUT    2
#define BETA    0.5f
#define THR     1.0f

typedef __attribute__((ext_vector_type(16))) __bf16 v16bf;
typedef __attribute__((ext_vector_type(8)))  __bf16 v8bf;
typedef __attribute__((ext_vector_type(8)))  float  v8f;
typedef __attribute__((ext_vector_type(4)))  unsigned int u32x4;
typedef __attribute__((ext_vector_type(8)))  int i32x8;
typedef __attribute__((ext_vector_type(4)))  int i32x4;

#if defined(__has_builtin)
#if __has_builtin(__builtin_amdgcn_tensor_load_to_lds) && __has_builtin(__builtin_amdgcn_s_wait_tensorcnt)
#define USE_TDM 1
#endif
#endif

// ---------------- workspace layout (bytes) ----------------
static constexpr size_t WS_W2BF = 0;                                 // bf16 [NHP][NHP], row n = W2[n][:]
static constexpr size_t WS_S1   = WS_W2BF + (size_t)NHP * NHP * 2;   // bf16 [BATCH][NHP]
static constexpr size_t WS_S2   = WS_S1   + (size_t)BATCH * NHP * 2; // bf16 [BATCH][NHP]
static constexpr size_t WS_M1   = WS_S2   + (size_t)BATCH * NHP * 2; // f32  [BATCH][NHP]
static constexpr size_t WS_M2   = WS_M1   + (size_t)BATCH * NHP * 4; // f32  [BATCH][NHP]
static constexpr size_t WS_M3   = WS_M2   + (size_t)BATCH * NHP * 4; // f32  [BATCH][NOUT]

// ---------------- init: bf16 weight prep + state zeroing (runs every call) ----------------
__global__ void snn_init_kernel(const float* __restrict__ W2, char* __restrict__ ws) {
    __bf16* w2bf = (__bf16*)(ws + WS_W2BF);
    float*  m1   = (float*)(ws + WS_M1);
    float*  m2   = (float*)(ws + WS_M2);
    float*  m3   = (float*)(ws + WS_M3);
    int tid = blockIdx.x * blockDim.x + threadIdx.x;
    int stride = gridDim.x * blockDim.x;

    for (int j = tid; j < NHP * NHP; j += stride) {
        int n = j >> 9, k = j & (NHP - 1);
        float v = (n < NH && k < NH) ? W2[n * NH + k] : 0.0f;
        w2bf[j] = (__bf16)v;
    }
    for (int j = tid; j < BATCH * NHP; j += stride) { m1[j] = 0.0f; m2[j] = 0.0f; }
    for (int j = tid; j < BATCH * NOUT; j += stride) m3[j] = 0.0f;
}

#ifdef USE_TDM
// TDM: async-load one [128 rows x 32 bf16] K-slab of the spike matrix into LDS.
// D# per CDNA5 ISA: group0 {count=1, lds_addr, global_addr, type=2},
// group1 {mask=0, data_size=1(2B), tensor_dim0=32, tensor_dim1=128, tile_dim0=32,
//         tile_dim1=128, tensor_dim0_stride=512}. Groups 2/3 zero (<=2D tile).
__device__ __forceinline__ void tdm_load_slab(const __bf16* gsrc, const __bf16* lds_dst) {
    unsigned long long ga = (unsigned long long)(uintptr_t)gsrc;
    unsigned lds = (unsigned)(uintptr_t)lds_dst;   // generic LDS ptr: low 32 bits = LDS byte addr
    u32x4 g0 = { 1u,                                           // count=1, user descriptor
                 lds,                                          // lds_addr
                 (unsigned)(ga & 0xFFFFFFFFu),                 // global_addr[31:0]
                 (unsigned)((ga >> 32) & 0x1FFFFFFu) | (2u << 30) }; // ga[56:32] | type=2
    i32x8 g1 = { (int)(1u << 16),      // workgroup_mask=0 | data_size=1 (2 bytes)
                 (int)(32u << 16),     // tensor_dim0[15:0]=32 at bits[63:48]
                 (int)(128u << 16),    // tensor_dim0[31:16]=0 | tensor_dim1[15:0]=128
                 (int)(32u << 16),     // tensor_dim1[31:16]=0 | tile_dim0=32
                 (int)128,             // tile_dim1=128 | tile_dim2=0
                 (int)512,             // tensor_dim0_stride[31:0]=512
                 0, 0 };               // stride hi / tensor_dim1_stride = 0
    i32x4 z4 = { 0, 0, 0, 0 };
#if defined(__clang_major__) && (__clang_major__ >= 23)
    i32x8 z8 = { 0, 0, 0, 0, 0, 0, 0, 0 };
    __builtin_amdgcn_tensor_load_to_lds(g0, g1, z4, z4, z8, 0);
#else
    __builtin_amdgcn_tensor_load_to_lds(g0, g1, z4, z4, 0);
#endif
}
#endif

// ---------------- persistent single-workgroup SNN kernel ----------------
// 512 threads = 16 wave32 waves on one WGP (4 waves/SIMD -> ample VGPR budget, no spills).
__global__ void __launch_bounds__(512, 1)
snn_run_kernel(const float* __restrict__ x,    // [T][B][NIN]
               const float* __restrict__ W1,   // [NH][NIN]
               const float* __restrict__ b1,   // [NH]
               const float* __restrict__ b2,   // [NH]
               const float* __restrict__ Wout, // [NOUT][NH]
               const float* __restrict__ bout, // [NOUT]
               char* __restrict__ ws,
               float* __restrict__ out)        // spk[T][B][2] then mem[T][B][2]
{
    __shared__ float red[512];
#ifdef USE_TDM
    __shared__ __bf16 ldsA[2][BATCH * 32];     // double-buffered A K-slabs (2 x 8KB)
#endif

    __bf16*       s1 = (__bf16*)(ws + WS_S1);
    __bf16*       s2 = (__bf16*)(ws + WS_S2);
    const __bf16* w2 = (const __bf16*)(ws + WS_W2BF);
    float*        m1 = (float*)(ws + WS_M1);
    float*        m2 = (float*)(ws + WS_M2);
    float*        m3 = (float*)(ws + WS_M3);

    const int tid   = threadIdx.x;
    const int lane  = tid & 31;        // wave32
    const int wave  = tid >> 5;        // 0..15
    const int lhalf = lane >> 4;       // half-wave
    const int l15   = lane & 15;

    // layer-2 tiling: C [128][512] -> 8 M-tiles x 32 N-tiles of 16x16 = 256 tiles.
    // wave w: M-tile pair {2mp, 2mp+1}; N-quadrant nq; two passes of 4 N-tiles each.
    const int mp = wave >> 2;          // 0..3
    const int nq = wave & 3;           // 0..3
    const int ar0 = (2 * mp + 0) * 16 + l15;   // A row (batch) for m-tile 0
    const int ar1 = (2 * mp + 1) * 16 + l15;   // A row for m-tile 1

    // layer-3 split: 256 (b,o) pairs x 2 K-halves
    const int p  = tid >> 1;           // 0..255
    const int q  = tid & 1;            // K half
    const int cb = p >> 1, co = p & 1;

    for (int t = 0; t < T_STEPS; ++t) {
        // ======== Phase A: layer 1 (exact fp32, K=6) + LIF1 -> bf16 spikes ========
        const float* xt = x + (size_t)t * (BATCH * NIN);
        for (int idx = tid; idx < BATCH * NHP; idx += 512) {
            int b = idx >> 9, n = idx & (NHP - 1);
            float cur = 0.0f;
            if (n < NH) {
                const float* w  = W1 + n * NIN;
                const float* xr = xt + b * NIN;
                cur = b1[n];
#pragma unroll
                for (int i2 = 0; i2 < NIN; ++i2) cur += xr[i2] * w[i2];
            }
            float mm  = m1[idx];
            float rst = (mm > THR) ? THR : 0.0f;
            float mn  = BETA * mm + cur - rst;
            m1[idx]   = mn;
            s1[idx]   = (__bf16)((mn > THR) ? 1.0f : 0.0f);
        }
        __threadfence_block();
        __syncthreads();

        // ======== Phase B: layer 2 via v_wmma_f32_16x16x32_bf16 + LIF2 ========
        for (int np = 0; np < 2; ++np) {
            const int tnb = (nq * 8 + np * 4) * 16;   // first output column of this pass
            // per-lane 32-bit byte offsets into w2 (saddr-friendly addressing)
            unsigned boff[4];
#pragma unroll
            for (int ni = 0; ni < 4; ++ni)
                boff[ni] = (unsigned)(((tnb + ni * 16 + l15) * NHP + lhalf * 8) * 2);

            v8f acc[2][4];
#pragma unroll
            for (int m = 0; m < 2; ++m)
#pragma unroll
                for (int ni = 0; ni < 4; ++ni)
                    acc[m][ni] = (v8f){0.f, 0.f, 0.f, 0.f, 0.f, 0.f, 0.f, 0.f};

#ifdef USE_TDM
            if (wave == 0) tdm_load_slab(s1, &ldsA[0][0]);   // prefetch K-chunk 0
#endif
            for (int kc = 0; kc < NHP / 32; ++kc) {
                const int kb = kc * 32;
#ifdef USE_TDM
                if (wave == 0) __builtin_amdgcn_s_wait_tensorcnt((short)0); // chunk kc landed
                __syncthreads();                                            // publish slab; retire old buffer
                if (wave == 0 && kc + 1 < NHP / 32)
                    tdm_load_slab(s1 + (kc + 1) * 32, &ldsA[(kc + 1) & 1][0]); // overlap next DMA
                const __bf16* sl = &ldsA[kc & 1][0];
                v8bf a0lo = *(const v8bf*)(sl + ar0 * 32 + lhalf * 8);
                v8bf a0hi = *(const v8bf*)(sl + ar0 * 32 + lhalf * 8 + 16);
                v8bf a1lo = *(const v8bf*)(sl + ar1 * 32 + lhalf * 8);
                v8bf a1hi = *(const v8bf*)(sl + ar1 * 32 + lhalf * 8 + 16);
#else
                const __bf16* arp0 = s1 + (size_t)ar0 * NHP + kb + lhalf * 8;
                const __bf16* arp1 = s1 + (size_t)ar1 * NHP + kb + lhalf * 8;
                v8bf a0lo = *(const v8bf*)(arp0);
                v8bf a0hi = *(const v8bf*)(arp0 + 16);
                v8bf a1lo = *(const v8bf*)(arp1);
                v8bf a1hi = *(const v8bf*)(arp1 + 16);
#endif
                v16bf af0 = __builtin_shufflevector(a0lo, a0hi, 0,1,2,3,4,5,6,7,8,9,10,11,12,13,14,15);
                v16bf af1 = __builtin_shufflevector(a1lo, a1hi, 0,1,2,3,4,5,6,7,8,9,10,11,12,13,14,15);

                v16bf bf[4];
#pragma unroll
                for (int ni = 0; ni < 4; ++ni)
                    bf[ni] = __builtin_shufflevector(*(const v8bf*)((const char*)w2 + boff[ni] + kb * 2), *(const v8bf*)((const char*)w2 + boff[ni] + kb * 2 + 32), 0,1,2,3,4,5,6,7,8,9,10,11,12,13,14,15);

#pragma unroll
                for (int ni = 0; ni < 4; ++ni) {
                    acc[0][ni] = __builtin_amdgcn_wmma_f32_16x16x32_bf16(
                        false, af0, false, bf[ni], (short)0, acc[0][ni], false, false);
                    acc[1][ni] = __builtin_amdgcn_wmma_f32_16x16x32_bf16(
                        false, af1, false, bf[ni], (short)0, acc[1][ni], false, false);
                }
                if (kc + 1 < NHP / 32)
                    __builtin_prefetch((const char*)w2 + boff[0] + kb * 2 + 64, 0, 1);
            }

            // LIF2 epilogue; C layout: VGPR r -> M = r + 8*lhalf, N = l15
#pragma unroll
            for (int m = 0; m < 2; ++m) {
#pragma unroll
                for (int ni = 0; ni < 4; ++ni) {
                    int n = tnb + ni * 16 + l15;
                    float bias = (n < NH) ? b2[n] : 0.0f;
#pragma unroll
                    for (int r = 0; r < 8; ++r) {
                        int b = (2 * mp + m) * 16 + lhalf * 8 + r;
                        size_t idx2 = (size_t)b * NHP + n;
                        float cur = acc[m][ni][r] + bias;
                        float mm  = m2[idx2];
                        float rst = (mm > THR) ? THR : 0.0f;
                        float mn  = BETA * mm + cur - rst;
                        m2[idx2]  = mn;
                        s2[idx2]  = (__bf16)((mn > THR) ? 1.0f : 0.0f);
                    }
                }
            }
        }
        __threadfence_block();
        __syncthreads();

        // ======== Phase C: layer 3 (exact fp32, N=2) + LIF3 + record ========
        {
            float sum = 0.0f;
            const __bf16* s2r = s2 + (size_t)cb * NHP;
            const float*  wr  = Wout + co * NH;
            const int k0 = q * 250, k1 = k0 + 250;   // 2*250 == NH
            for (int k = k0; k < k1; ++k) sum += (float)s2r[k] * wr[k];
            red[tid] = sum;
        }
        __syncthreads();
        if (q == 0) {
            float cur = red[tid] + red[tid + 1] + bout[co];
            int   mi  = cb * NOUT + co;
            float mm  = m3[mi];
            float rst = (mm > THR) ? THR : 0.0f;
            float mn  = BETA * mm + cur - rst;
            m3[mi]    = mn;
            float sp  = (mn > THR) ? 1.0f : 0.0f;
            size_t ob = (size_t)t * (BATCH * NOUT) + mi;
            out[ob] = sp;                                        // spk_rec
            out[(size_t)T_STEPS * BATCH * NOUT + ob] = mn;       // mem_rec
        }
    }
}

// ---------------- launcher ----------------
extern "C" void kernel_launch(void* const* d_in, const int* in_sizes, int n_in,
                              void* d_out, int out_size, void* d_ws, size_t ws_size,
                              hipStream_t stream) {
    const float* x    = (const float*)d_in[0];
    const float* W1   = (const float*)d_in[1];
    const float* b1   = (const float*)d_in[2];
    const float* W2   = (const float*)d_in[3];
    const float* b2   = (const float*)d_in[4];
    const float* Wout = (const float*)d_in[5];
    const float* bout = (const float*)d_in[6];
    float* out = (float*)d_out;
    char*  ws  = (char*)d_ws;

    // Re-initialize bf16 weights + membrane state every call (deterministic replay).
    snn_init_kernel<<<512, 256, 0, stream>>>(W2, ws);
    // Persistent single-workgroup kernel: recurrence over 500 steps is sync-latency
    // bound, so keep it on one WGP with hardware barriers + TDM double buffering.
    snn_run_kernel<<<1, 512, 0, stream>>>(x, W1, b1, b2, Wout, bout, ws, out);
}
